// ContextCluster_85409719648518
// MI455X (gfx1250) — compile-verified, matrix operand order checked by reading, not run on hardware
//
#include <hip/hip_runtime.h>
#include <math.h>

// ---------------------------------------------------------------------------
// ContextCluster for MI455X (gfx1250), wave32 + WMMA bf16 (f32 accumulate)
// + Tensor Data Mover for weight staging.
//
// Roofline: ~16.3 GFLOP of GEMM, ~360 MB minimal HBM traffic => memory bound
// (~15us @ 23.3 TB/s). Never materialize the 170MB f/value tensors; exploit
// linearity (centers = Wf @ pool(x) + bf); per-tile WMMA in LDS; spill only
// sim(f32)+idx(u8) per point; second WMMA pass for projection.
// 128-pt tiles: 145.5KB LDS/block -> 2 blocks/WGP (291KB of 320KB).
// ---------------------------------------------------------------------------

typedef __attribute__((ext_vector_type(16))) __bf16 v16bf;
typedef __attribute__((ext_vector_type(8)))  float  v8f;
typedef __attribute__((ext_vector_type(4)))  unsigned int u32x4;
typedef __attribute__((ext_vector_type(8)))  unsigned int u32x8;

#define NPTS 13824   // 24^3 points per folded sub-problem
#define TILE 128     // points per block tile
#define NTILES 108   // 13824 / 128

// K index for lane/element of a 16x32 bf16 WMMA A fragment (ISA 7.12.2),
// and (by symmetric assumption) for the 32x16 B fragment.
__device__ __forceinline__ int kmap(int lane, int i) {
    return ((lane & 16) ? 8 : 0) + ((i & 8) ? 16 : 0) + (((i >> 1) & 3) << 1) + (i & 1);
}

// Folded-point -> global 48^3 coordinates. Octant o = fw*4+fh*2+fd.
__device__ __forceinline__ void ptcoord(int n, int o, int &w, int &h, int &d) {
    int wp = n / 576, hp = (n / 24) % 24, dp = n % 24;
    w = ((o >> 2) & 1) * 24 + wp;
    h = ((o >> 1) & 1) * 24 + hp;
    d = (o & 1) * 24 + dp;
}

// TDM: 1-D contiguous global->LDS copy of `ndw` dwords (ndw <= 65535).
// D# per CDNA5 ISA ch.8: group0 {count=1, lds_addr, global_addr, type=2},
// group1 {data_size=4B, tensor_dim0=ndw, tensor_dim1=1, tile_dim0=ndw,
// tile_dim1=1, stride0=ndw}. Issued by one wave; completion via TENSORcnt.
__device__ __forceinline__ void tdm_load_1d(const void* gptr, void* lptr,
                                            unsigned int ndw) {
    unsigned long long ga = (unsigned long long)gptr;
    unsigned int lds = (unsigned int)(unsigned long long)lptr;  // addr[31:0] = LDS offset
    u32x4 g0;
    g0[0] = 1u;                                   // count=1 (valid), user mode
    g0[1] = lds;                                  // lds_addr
    g0[2] = (unsigned int)ga;                     // global_addr[31:0]
    g0[3] = (unsigned int)(ga >> 32) | (2u << 30);// global_addr[56:32] | type=2
    u32x8 g1;
    g1[0] = 2u << 16;                             // data_size=2 (4 bytes)
    g1[1] = (ndw & 0xFFFFu) << 16;                // tensor_dim0[15:0]
    g1[2] = (ndw >> 16) | (1u << 16);             // tensor_dim0[31:16], tensor_dim1=1
    g1[3] = ndw << 16;                            // tile_dim0 = ndw
    g1[4] = 1u;                                   // tile_dim1=1, tile_dim2=0
    g1[5] = ndw;                                  // tensor_dim0_stride[31:0]
    g1[6] = 0u;
    g1[7] = 0u;
    asm volatile("tensor_load_to_lds %0, %1" :: "s"(g0), "s"(g1) : "memory");
}

// ---- K0: pre-convert weights to bf16, stack biases --------------------------
__global__ void prep_kernel(const float* Wf, const float* Wv, const float* bf,
                            const float* bv, const float* Wp,
                            __bf16* Wfv, __bf16* Wp16, float* biasFV) {
    int t = blockIdx.x * blockDim.x + threadIdx.x;
    int stride = gridDim.x * blockDim.x;
    for (int j = t; j < 192 * 64; j += stride) {
        int row = j >> 6, k = j & 63;
        float v = (row < 96) ? Wf[row * 64 + k] : Wv[(row - 96) * 64 + k];
        Wfv[j] = (__bf16)v;
    }
    for (int j = t; j < 64 * 96; j += stride) Wp16[j] = (__bf16)Wp[j];
    for (int j = t; j < 192; j += stride) biasFV[j] = (j < 96) ? bf[j] : bv[j - 96];
}

// ---- K0b: zero the atomic aggregation buffers (every launch) ----------------
__global__ void zero_kernel(float* p, int n) {
    int t = blockIdx.x * blockDim.x + threadIdx.x;
    if (t < n) p[t] = 0.f;
}

// ---- K1: pool x over each 12^3 cell -> xpool[4][64][64] ---------------------
__global__ void pool_kernel(const float* x, float* xpool) {
    int cell = blockIdx.x;   // 0..63 : cw*16 + ch*4 + cd
    int b0 = blockIdx.y;     // 0..3
    int t = threadIdx.x;     // 256
    int ch = t >> 2, sub = t & 3;
    int cw = cell >> 4, chh = (cell >> 2) & 3, cd = cell & 3;
    const float* base = x + ((((size_t)(b0 * 64 + ch) * 48 + cw * 12) * 48 + chh * 12) * 48 + cd * 12);
    float s = 0.f;
    for (int i = 0; i < 432; ++i) {
        int e = sub * 432 + i;
        int wi = e / 144, hi = (e / 12) % 12, di = e % 12;
        s += base[(wi * 48 + hi) * 48 + di];
    }
    __shared__ float red[256];
    red[t] = s;
    __syncthreads();
    if (sub == 0) {
        float m = (red[t] + red[t + 1] + red[t + 2] + red[t + 3]) * (1.f / 1728.f);
        xpool[(b0 * 64 + ch) * 64 + cell] = m;
    }
}

// ---- K2: centers = Wf@xpool+bf (L2-normalized), value_centers = Wv@xpool+bv -
__global__ void centers_kernel(const float* xpool, const float* Wf, const float* Wv,
                               const float* bf, const float* bv,
                               float* cenN, float* vcent) {
    int bm = blockIdx.x;                 // 0..1023 = bfold*8 + m
    int bfold = bm >> 3, m = bm & 7;
    int b0 = bfold >> 5, e = (bfold >> 3) & 3, o = bfold & 7;
    int cw = ((o >> 2) & 1) * 2 + ((m >> 2) & 1);
    int chh = ((o >> 1) & 1) * 2 + ((m >> 1) & 1);
    int cd = (o & 1) * 2 + (m & 1);
    int cell = cw * 16 + chh * 4 + cd;
    __shared__ float xp[64];
    __shared__ float fc[24];
    int t = threadIdx.x;                 // 64 threads
    xp[t] = xpool[(b0 * 64 + t) * 64 + cell];
    __syncthreads();
    if (t < 48) {
        int c = t % 24;
        int row = e * 24 + c;
        const float* W = (t < 24) ? Wf : Wv;
        float acc = (t < 24) ? bf[row] : bv[row];
        for (int k = 0; k < 64; ++k) acc += W[row * 64 + k] * xp[k];
        if (t < 24) fc[c] = acc;
        else        vcent[(bfold * 8 + m) * 24 + c] = acc;
    }
    __syncthreads();
    if (t < 24) {
        float n2 = 0.f;
        for (int i = 0; i < 24; ++i) n2 += fc[i] * fc[i];
        cenN[(bfold * 8 + m) * 24 + t] = fc[t] / fmaxf(sqrtf(n2), 1e-12f);
    }
}

// ---- K3: main fused pass: WMMA f/value GEMM + cluster assign + aggregate ----
__global__ void __launch_bounds__(256) main_kernel(
    const float* x, const __bf16* Wfv, const float* biasFV,
    const float* cenN, const float* alpha, const float* beta,
    float* aggv, float* aggs, float* simsG, unsigned char* idxG) {
    __shared__ __bf16 Alds[192 * 64];      // stacked [f;v] weights, [row][k]  24.0KB
    __shared__ __bf16 Blds[64 * TILE];     // x tile (bf16), [k][p]            16.0KB
    __shared__ float  Clds[192 * TILE];    // f/value results, [row][p]        96.0KB
    __shared__ float  cen[4 * 8 * 24];     // normalized centers for this (b0,o)
    __shared__ float  aggL[4 * 8 * 25];    // block-local aggregation buckets
    int t = threadIdx.x;
    int tileI = blockIdx.x, o = blockIdx.y, b0 = blockIdx.z;
    int n0 = tileI * TILE;
    // stage A (bf16 weights) via Tensor Data Mover (wave 0 issues + waits)
    if (t < 32) {
        tdm_load_1d(Wfv, Alds, 192 * 64 / 2);
        __builtin_amdgcn_s_wait_tensorcnt(0);
    }
    for (int i = 0; i < 32; ++i) {         // stage + convert x tile
        int flat = i * 256 + t;
        int ch = flat >> 7, p = flat & 127;
        int n = n0 + p, w, h, d;
        ptcoord(n, o, w, h, d);
        Blds[ch * TILE + p] =
            (__bf16)x[(((size_t)(b0 * 64 + ch) * 48 + w) * 48 + h) * 48 + d];
    }
    for (int j = t; j < 4 * 8 * 24; j += 256) {
        int e = j / 192;
        cen[j] = cenN[(size_t)(((b0 * 4 + e) * 8 + o) * 8) * 24 + (j % 192)];
    }
    for (int j = t; j < 4 * 8 * 25; j += 256) aggL[j] = 0.f;
    __syncthreads();

    // WMMA: 192 rows x 128 pts, K=64. 8 waves: Ntile = wave, 12 Mtiles each.
    int wave = t >> 5, lane = t & 31;
    int nt = wave;
    v16bf bfrag[2];
    for (int ks = 0; ks < 2; ++ks) {
#pragma unroll
        for (int i = 0; i < 16; ++i)
            bfrag[ks][i] = Blds[(ks * 32 + kmap(lane, i)) * TILE + nt * 16 + (lane & 15)];
    }
    for (int mt = 0; mt < 12; ++mt) {
        v8f acc;
#pragma unroll
        for (int j = 0; j < 8; ++j)        // bias folded into accumulator init
            acc[j] = biasFV[mt * 16 + j + ((lane & 16) ? 8 : 0)];
        for (int ks = 0; ks < 2; ++ks) {
            v16bf afrag;
#pragma unroll
            for (int i = 0; i < 16; ++i)
                afrag[i] = Alds[(mt * 16 + (lane & 15)) * 64 + ks * 32 + kmap(lane, i)];
            acc = __builtin_amdgcn_wmma_f32_16x16x32_bf16(
                false, afrag, false, bfrag[ks], (short)0, acc, false, false);
        }
#pragma unroll
        for (int j = 0; j < 8; ++j)
            Clds[(mt * 16 + j + ((lane & 16) ? 8 : 0)) * TILE + nt * 16 + (lane & 15)] =
                acc[j];
    }
    __syncthreads();

    // per-(head,point): normalize, cosine vs 8 centers, sigmoid, argmax, agg
    float sa = alpha[0], sb = beta[0];
    for (int jj = 0; jj < 2; ++jj) {
        int job = jj * 256 + t;            // 512 jobs = 4 heads x 128 points
        int e = job >> 7, p = job & 127;
        float fv[24], vv[24];
        float n2 = 0.f;
#pragma unroll
        for (int c = 0; c < 24; ++c) {
            fv[c] = Clds[(e * 24 + c) * TILE + p];
            n2 += fv[c] * fv[c];
            vv[c] = Clds[(96 + e * 24 + c) * TILE + p];
        }
        float inv = 1.f / fmaxf(sqrtf(n2), 1e-12f);
        float bestS = -1.f; int bestM = 0;
        for (int m = 0; m < 8; ++m) {
            float dot = 0.f;
#pragma unroll
            for (int c = 0; c < 24; ++c) dot += fv[c] * cen[(e * 8 + m) * 24 + c];
            float sim = 1.f / (1.f + expf(-(sb + sa * dot * inv)));
            if (sim > bestS) { bestS = sim; bestM = m; }
        }
#pragma unroll
        for (int c = 0; c < 24; ++c)
            atomicAdd(&aggL[(e * 8 + bestM) * 25 + c], bestS * vv[c]);
        atomicAdd(&aggL[(e * 8 + bestM) * 25 + 24], bestS);
        int bi = (b0 * 4 + e) * 8 + o;
        int n = n0 + p;
        simsG[(size_t)bi * NPTS + n] = bestS;
        idxG[(size_t)bi * NPTS + n] = (unsigned char)bestM;
    }
    __syncthreads();
    for (int j = t; j < 4 * 8 * 25; j += 256) {   // flush to global atomics
        int em = j / 25, c = j % 25;
        int e = em >> 3, m = em & 7;
        int bi = (b0 * 4 + e) * 8 + o;
        if (c < 24) atomicAdd(&aggv[(bi * 8 + m) * 24 + c], aggL[j]);
        else        atomicAdd(&aggs[bi * 8 + m], aggL[j]);
    }
}

// ---- K4: agg = (sum(sim*v) + value_center) / (sum(sim) + 1) -----------------
__global__ void finalize_kernel(const float* aggv, const float* aggs,
                                const float* vcent, float* aggfin) {
    int idx = blockIdx.x * 256 + threadIdx.x;
    if (idx >= 128 * 8 * 24) return;
    aggfin[idx] = (aggv[idx] + vcent[idx]) / (aggs[idx / 24] + 1.f);
}

// ---- K5: dispatch (sim * agg[idx]) + output projection (WMMA 64x96) ---------
__global__ void __launch_bounds__(256) project_kernel(
    const __bf16* Wp16, const float* bp, const float* aggfin,
    const float* simsG, const unsigned char* idxG, float* out) {
    __shared__ __bf16 Awp[64 * 96];        // Wp bf16 [row][k]   12KB
    __shared__ __bf16 Gl[96 * TILE];       // dispatched features, [k][p] 24KB
    __shared__ float  aggLds[4 * 8 * 24];
    int t = threadIdx.x;
    int tileI = blockIdx.x, o = blockIdx.y, b0 = blockIdx.z;
    int n0 = tileI * TILE;
    if (t < 32) {                          // stage Wp via TDM
        tdm_load_1d(Wp16, Awp, 64 * 96 / 2);
        __builtin_amdgcn_s_wait_tensorcnt(0);
    }
    for (int j = t; j < 4 * 8 * 24; j += 256) {
        int e = j / 192;
        aggLds[j] = aggfin[(size_t)(((b0 * 4 + e) * 8 + o) * 8) * 24 + (j % 192)];
    }
    __syncthreads();
    for (int jj = 0; jj < 2; ++jj) {       // build dispatched features
        int job = jj * 256 + t;
        int e = job >> 7, p = job & 127;
        int bi = (b0 * 4 + e) * 8 + o;
        int n = n0 + p;
        float sim = simsG[(size_t)bi * NPTS + n];
        int m = idxG[(size_t)bi * NPTS + n];
#pragma unroll
        for (int c = 0; c < 24; ++c)
            Gl[(e * 24 + c) * TILE + p] = (__bf16)(sim * aggLds[(e * 8 + m) * 24 + c]);
    }
    __syncthreads();
    int wave = t >> 5, lane = t & 31;
    for (int q = 0; q < 4; ++q) {          // 32 (mt,nt) pairs over 8 waves
        int pr = wave * 4 + q;
        int mt = pr >> 3, ntl = pr & 7;
        v8f acc;
#pragma unroll
        for (int j = 0; j < 8; ++j)
            acc[j] = bp[mt * 16 + j + ((lane & 16) ? 8 : 0)];
        for (int ks = 0; ks < 3; ++ks) {
            v16bf afrag, bfrag;
#pragma unroll
            for (int i = 0; i < 16; ++i) {
                int k = ks * 32 + kmap(lane, i);
                afrag[i] = Awp[(mt * 16 + (lane & 15)) * 96 + k];
                bfrag[i] = Gl[k * TILE + ntl * 16 + (lane & 15)];
            }
            acc = __builtin_amdgcn_wmma_f32_16x16x32_bf16(
                false, afrag, false, bfrag, (short)0, acc, false, false);
        }
#pragma unroll
        for (int j = 0; j < 8; ++j) {
            int ch = mt * 16 + j + ((lane & 16) ? 8 : 0);
            int p = ntl * 16 + (lane & 15);
            int n = n0 + p, w, h, d;
            ptcoord(n, o, w, h, d);
            out[(((size_t)(b0 * 64 + ch) * 48 + w) * 48 + h) * 48 + d] = acc[j];
        }
    }
}

// ---------------------------------------------------------------------------
extern "C" void kernel_launch(void* const* d_in, const int* in_sizes, int n_in,
                              void* d_out, int out_size, void* d_ws, size_t ws_size,
                              hipStream_t stream) {
    (void)in_sizes; (void)n_in; (void)out_size; (void)ws_size;
    const float* x     = (const float*)d_in[0];
    const float* Wf    = (const float*)d_in[1];
    const float* bf    = (const float*)d_in[2];
    const float* Wv    = (const float*)d_in[3];
    const float* bv    = (const float*)d_in[4];
    const float* Wp    = (const float*)d_in[5];
    const float* bp    = (const float*)d_in[6];
    const float* alpha = (const float*)d_in[7];
    const float* beta  = (const float*)d_in[8];
    float* out = (float*)d_out;

    char* ws = (char*)d_ws;
    float*         xpool  = (float*)(ws + 0);          //  65536 B
    float*         cenN   = (float*)(ws + 65536);      //  98304 B
    float*         vcent  = (float*)(ws + 163840);     //  98304 B
    __bf16*        Wfv    = (__bf16*)(ws + 262144);    //  24576 B
    __bf16*        Wp16   = (__bf16*)(ws + 286720);    //  12288 B
    float*         biasFV = (float*)(ws + 299008);     //    768 B (padded)
    float*         aggv   = (float*)(ws + 300032);     //  98304 B
    float*         aggs   = (float*)(ws + 398336);     //   4096 B (contiguous after aggv)
    float*         aggfin = (float*)(ws + 402432);     //  98304 B
    float*         simsG  = (float*)(ws + 500736);     // 7077888 B
    unsigned char* idxG   = (unsigned char*)(ws + 7578624); // 1769472 B  (total ~9.35 MB)

    prep_kernel<<<dim3(8), 256, 0, stream>>>(Wf, Wv, bf, bv, Wp, Wfv, Wp16, biasFV);
    zero_kernel<<<dim3(100), 256, 0, stream>>>(aggv, 25600);   // aggv + aggs
    pool_kernel<<<dim3(64, 4), 256, 0, stream>>>(x, xpool);
    centers_kernel<<<dim3(1024), 64, 0, stream>>>(xpool, Wf, Wv, bf, bv, cenN, vcent);
    main_kernel<<<dim3(NTILES, 8, 4), 256, 0, stream>>>(
        x, Wfv, biasFV, cenN, alpha, beta, aggv, aggs, simsG, idxG);
    finalize_kernel<<<dim3(96), 256, 0, stream>>>(aggv, aggs, vcent, aggfin);
    project_kernel<<<dim3(NTILES, 8, 4), 256, 0, stream>>>(
        Wp16, bp, aggfin, simsG, idxG, out);
}
